// SelfAttentionBlock_52862457479216
// MI455X (gfx1250) — matrix operand order repaired and checked by reading, not compiled, not run on hardware
//
#include <hip/hip_runtime.h>
#include <hip/hip_bf16.h>

typedef __attribute__((ext_vector_type(16))) _Float16 v16h;
typedef __attribute__((ext_vector_type(8)))  _Float16 v8h;
typedef __attribute__((ext_vector_type(8)))  float    v8f;

#define BATCH   8
#define CH      512
#define NGROUPS 32
#define CG      16      // CH / NGROUPS
#define HEADS   8
#define HD      64
#define NSP     1024    // H*W

// ---------------------------------------------------------------------------
// Fragment loaders matching the CDNA5 WMMA VGPR layout tables.
// A-matrix 16x32 f16: lane holds row M = lane&15; halves 0..7 = K hi*8+0..7,
// halves 8..15 = K 16+hi*8+0..7  -> two contiguous b128 loads.
// B-matrix 32x16 f16: lane holds col N = lane&15; halves 0..7 = K hi*8+0..7,
// halves 8..15 = K 16+hi*8+0..7  -> two contiguous b128 loads.
// ---------------------------------------------------------------------------
__device__ __forceinline__ v16h load_a(const _Float16* p, int hi) {
    v8h lo = *(const v8h*)(p + hi * 8);
    v8h hh = *(const v8h*)(p + 16 + hi * 8);
    return __builtin_shufflevector(lo, hh, 0, 1, 2, 3, 4, 5, 6, 7,
                                   8, 9, 10, 11, 12, 13, 14, 15);
}
__device__ __forceinline__ v16h load_b(const _Float16* p, int hi) {
    v8h lo = *(const v8h*)(p + hi * 8);
    v8h hh = *(const v8h*)(p + 16 + hi * 8);
    return __builtin_shufflevector(lo, hh, 0, 1, 2, 3, 4, 5, 6, 7,
                                   8, 9, 10, 11, 12, 13, 14, 15);
}
__device__ __forceinline__ v8f wmma16(v16h a, v16h b, v8f c) {
    return __builtin_amdgcn_wmma_f32_16x16x32_f16(false, a, false, b,
                                                  (short)0, c, false, false);
}

// ---------------------------------------------------------------------------
// Kernel 0: f32 -> f16 conversion (weights)
// ---------------------------------------------------------------------------
__global__ __launch_bounds__(256) void cvt_kernel(
    const float* __restrict__ src, _Float16* __restrict__ dst, int n) {
    int i = blockIdx.x * 256 + threadIdx.x;
    if (i < n) dst[i] = (_Float16)src[i];
}

// ---------------------------------------------------------------------------
// Kernel 1: GroupNorm -> f16 xT[b][n][c]   (K=c contiguous for WMMA loads)
// ---------------------------------------------------------------------------
__global__ __launch_bounds__(256) void gn_kernel(
    const float* __restrict__ x, const float* __restrict__ gamma,
    const float* __restrict__ beta, _Float16* __restrict__ xT) {
    int blk = blockIdx.x;            // b*NGROUPS + g
    int b = blk >> 5, g = blk & 31;
    const float* xb = x + (size_t)(b * CH + g * CG) * NSP;

    float s = 0.f, ss = 0.f;
    for (int i = threadIdx.x; i < CG * NSP; i += 256) {
        float v = xb[i];
        s += v; ss += v * v;
    }
    __shared__ float red0[256];
    __shared__ float red1[256];
    red0[threadIdx.x] = s; red1[threadIdx.x] = ss;
    __syncthreads();
    for (int off = 128; off > 0; off >>= 1) {
        if ((int)threadIdx.x < off) {
            red0[threadIdx.x] += red0[threadIdx.x + off];
            red1[threadIdx.x] += red1[threadIdx.x + off];
        }
        __syncthreads();
    }
    const float invCnt = 1.f / (float)(CG * NSP);
    float mean = red0[0] * invCnt;
    float var  = red1[0] * invCnt - mean * mean;
    float rstd = rsqrtf(var + 1e-6f);

    _Float16* xo = xT + (size_t)b * NSP * CH;
    for (int i = threadIdx.x; i < CG * NSP; i += 256) {
        int c = g * CG + (i >> 10);
        int n = i & 1023;
        float v = (xb[i] - mean) * rstd * gamma[c] + beta[c];
        xo[(size_t)n * CH + c] = (_Float16)v;
    }
}

// ---------------------------------------------------------------------------
// Kernel 2: QKV GEMM. Two orientations (uniform per block):
//   Q/K: rows=n, cols=o  -> coalesced stores into q/k[bh][n][d]
//   V  : rows=o, cols=n  -> coalesced stores into v[bh][d][m]
// ---------------------------------------------------------------------------
__global__ __launch_bounds__(256) void qkv_kernel(
    const _Float16* __restrict__ xT,   // [b][n][c]
    const _Float16* __restrict__ w16,  // [o][c] f16
    const float* __restrict__ b_qkv,
    _Float16* __restrict__ qbuf,       // [bh][n][d]
    _Float16* __restrict__ kbuf,       // [bh][n][d]
    _Float16* __restrict__ vbuf) {     // [bh][d][m]
    int wave = threadIdx.x >> 5, lane = threadIdx.x & 31;
    int lh = lane & 15, hi = lane >> 4;
    int oBlk = blockIdx.x * 128;       // 12 blocks -> o in [0,1536)
    int nBlk = blockIdx.y * 64;
    int b = blockIdx.z;
    int part = oBlk >> 9;              // uniform per block
    const _Float16* xb = xT + (size_t)b * NSP * CH;

    v8f acc[4] = {};
    if (part < 2) {
        // ----- Q / K : transposed orientation -----
        int o64 = oBlk + (wave & 1) * 64;
        int nT  = nBlk + (wave >> 1) * 16;
        const _Float16* arow = xb + (size_t)(nT + lh) * CH;
        for (int k0 = 0; k0 < CH; k0 += 32) {
            v16h a = load_a(arow + k0, hi);
            #pragma unroll
            for (int t = 0; t < 4; ++t) {
                int o = o64 + t * 16 + lh;
                v16h bf = load_b(w16 + (size_t)o * CH + k0, hi);
                acc[t] = wmma16(a, bf, acc[t]);
            }
        }
        _Float16* dst = (part == 0) ? qbuf : kbuf;
        #pragma unroll
        for (int t = 0; t < 4; ++t) {
            int o = o64 + t * 16 + lh;
            float bias = b_qkv[o];
            int h = (o >> 6) & 7, d = o & 63;
            size_t bh = (size_t)b * HEADS + h;
            #pragma unroll
            for (int r = 0; r < 8; ++r) {
                int n = nT + r + 8 * hi;
                dst[(bh * NSP + n) * HD + d] = (_Float16)(acc[t][r] + bias);
            }
        }
    } else {
        // ----- V : normal orientation -----
        int oT = oBlk + wave * 16;
        const _Float16* arow = w16 + (size_t)(oT + lh) * CH;
        for (int k0 = 0; k0 < CH; k0 += 32) {
            v16h a = load_a(arow + k0, hi);
            #pragma unroll
            for (int t = 0; t < 4; ++t) {
                int n = nBlk + t * 16 + lh;
                v16h bf = load_b(xb + (size_t)n * CH + k0, hi);
                acc[t] = wmma16(a, bf, acc[t]);
            }
        }
        #pragma unroll
        for (int r = 0; r < 8; ++r) {
            int o = oT + r + 8 * hi;
            float bias = b_qkv[o];
            int h = (o >> 6) & 7, d = o & 63;
            size_t bh = (size_t)b * HEADS + h;
            #pragma unroll
            for (int t = 0; t < 4; ++t) {
                int n = nBlk + t * 16 + lh;
                vbuf[(bh * HD + d) * NSP + n] = (_Float16)(acc[t][r] + bias);
            }
        }
    }
}

// ---------------------------------------------------------------------------
// Kernel 3: flash-attention, one wave per 16-query tile per (b,h).
// ---------------------------------------------------------------------------
__global__ __launch_bounds__(256) void attn_kernel(
    const _Float16* __restrict__ qbuf,   // [bh][n][d]
    const _Float16* __restrict__ kbuf,   // [bh][m][d]
    const _Float16* __restrict__ vbuf,   // [bh][d][m]
    _Float16* __restrict__ attnT) {      // [b][n][c]
    __shared__ __align__(64) _Float16 ptile[8][16][32];  // per-wave P staging

    int wave = threadIdx.x >> 5, lane = threadIdx.x & 31;
    int lh = lane & 15, hi = lane >> 4;
    int gw = blockIdx.x * 8 + wave;   // 0..4095
    int qt = gw & 63;
    int bh = gw >> 6;
    int b = bh >> 3, h = bh & 7;
    int nBase = qt * 16;
    const float scale = 0.125f;       // hd^-0.5

    const _Float16* qp = qbuf + ((size_t)bh * NSP + nBase + lh) * HD;
    v16h aq0 = load_a(qp, hi);        // d 0..31
    v16h aq1 = load_a(qp + 32, hi);   // d 32..63

    float rmax[8], rsum[8];
    #pragma unroll
    for (int r = 0; r < 8; ++r) { rmax[r] = -1e30f; rsum[r] = 0.f; }
    v8f oacc[4] = {};

    const _Float16* kb = kbuf + (size_t)bh * NSP * HD;   // [m][d]
    const _Float16* vb = vbuf + (size_t)bh * HD * NSP;   // [d][m]

    for (int m0 = 0; m0 < NSP; m0 += 32) {
        if (m0 + 32 < NSP) {  // prefetch next K/V tiles (global_prefetch_b8)
            __builtin_prefetch(kb + (size_t)(m0 + 32 + lane) * HD, 0, 0);
            __builtin_prefetch(vb + (size_t)(lane * 2) * NSP + m0 + 32, 0, 0);
        }
        // S tile 16x32
        v8f sacc[2] = {};
        #pragma unroll
        for (int t = 0; t < 2; ++t) {
            const _Float16* kp = kb + (size_t)(m0 + t * 16 + lh) * HD;
            v16h bk0 = load_b(kp, hi);       // d = hi*8 + 0..7, 16 + hi*8 + 0..7
            v16h bk1 = load_b(kp + 32, hi);  // d = 32 + hi*8 + 0..7, 48 + hi*8 + 0..7
            sacc[t] = wmma16(aq0, bk0, sacc[t]);
            sacc[t] = wmma16(aq1, bk1, sacc[t]);
        }

        // online softmax; row = r + 8*hi, 16 lanes of a half share rows
        #pragma unroll
        for (int r = 0; r < 8; ++r) {
            float v0 = sacc[0][r] * scale;
            float v1 = sacc[1][r] * scale;
            float tm = fmaxf(v0, v1);
            tm = fmaxf(tm, __shfl_xor(tm, 1, 32));
            tm = fmaxf(tm, __shfl_xor(tm, 2, 32));
            tm = fmaxf(tm, __shfl_xor(tm, 4, 32));
            tm = fmaxf(tm, __shfl_xor(tm, 8, 32));
            float nm = fmaxf(rmax[r], tm);
            float corr = __expf(rmax[r] - nm);
            rmax[r] = nm;
            float p0 = __expf(v0 - nm);
            float p1 = __expf(v1 - nm);
            float ps = p0 + p1;
            ps += __shfl_xor(ps, 1, 32);
            ps += __shfl_xor(ps, 2, 32);
            ps += __shfl_xor(ps, 4, 32);
            ps += __shfl_xor(ps, 8, 32);
            rsum[r] = rsum[r] * corr + ps;
            #pragma unroll
            for (int c = 0; c < 4; ++c) oacc[c][r] *= corr;
            ptile[wave][r + 8 * hi][lh]      = (_Float16)p0;
            ptile[wave][r + 8 * hi][lh + 16] = (_Float16)p1;
        }

        // P back in A layout via LDS (2x ds_load_b128 per lane)
        v16h ap = load_a(&ptile[wave][lh][0], hi);

        // O += P * V
        #pragma unroll
        for (int c = 0; c < 4; ++c) {
            int d = c * 16 + lh;
            v16h bv = load_b(vb + (size_t)d * NSP + m0, hi);  // m contiguous
            oacc[c] = wmma16(ap, bv, oacc[c]);
        }
    }

    // normalize, store as attnT[b][n][c] (lane-contiguous along c)
    #pragma unroll
    for (int r = 0; r < 8; ++r) {
        float inv = 1.f / rsum[r];
        int n = nBase + r + 8 * hi;
        _Float16* dst = attnT + ((size_t)b * NSP + n) * CH + h * HD;
        #pragma unroll
        for (int c = 0; c < 4; ++c) {
            dst[c * 16 + lh] = (_Float16)(oacc[c][r] * inv);
        }
    }
}

// ---------------------------------------------------------------------------
// Kernel 4: proj GEMM + bias + residual -> f32 out[b][o][n]
// ---------------------------------------------------------------------------
__global__ __launch_bounds__(256) void proj_kernel(
    const _Float16* __restrict__ attnT,  // [b][n][c]
    const _Float16* __restrict__ w16,    // [o][c] f16
    const float* __restrict__ b_proj, const float* __restrict__ x,
    float* __restrict__ out) {
    int wave = threadIdx.x >> 5, lane = threadIdx.x & 31;
    int lh = lane & 15, hi = lane >> 4;
    int oT = blockIdx.x * 128 + wave * 16;
    int nBlk = blockIdx.y * 64;
    int b = blockIdx.z;
    const _Float16* ab = attnT + (size_t)b * NSP * CH;

    v8f acc[4] = {};
    const _Float16* arow = w16 + (size_t)(oT + lh) * CH;
    for (int k0 = 0; k0 < CH; k0 += 32) {
        v16h a = load_a(arow + k0, hi);
        #pragma unroll
        for (int t = 0; t < 4; ++t) {
            int n = nBlk + t * 16 + lh;
            v16h bf = load_b(ab + (size_t)n * CH + k0, hi);
            acc[t] = wmma16(a, bf, acc[t]);
        }
    }
    #pragma unroll
    for (int r = 0; r < 8; ++r) {
        int o = oT + r + 8 * hi;
        float bias = b_proj[o];
        #pragma unroll
        for (int t = 0; t < 4; ++t) {
            int n = nBlk + t * 16 + lh;
            size_t idx = ((size_t)b * CH + o) * NSP + n;
            out[idx] = acc[t][r] + bias + x[idx];
        }
    }
}

// ---------------------------------------------------------------------------
extern "C" void kernel_launch(void* const* d_in, const int* in_sizes, int n_in,
                              void* d_out, int out_size, void* d_ws, size_t ws_size,
                              hipStream_t stream) {
    const float* x      = (const float*)d_in[0];
    const float* gamma  = (const float*)d_in[1];
    const float* beta   = (const float*)d_in[2];
    const float* w_qkv  = (const float*)d_in[3];
    const float* b_qkv  = (const float*)d_in[4];
    const float* w_proj = (const float*)d_in[5];
    const float* b_proj = (const float*)d_in[6];
    float* out = (float*)d_out;

    char* ws = (char*)d_ws;
    const size_t plane = (size_t)BATCH * CH * NSP * sizeof(_Float16); // 8 MB
    _Float16* xT      = (_Float16*)(ws + 0 * plane);
    _Float16* qbuf    = (_Float16*)(ws + 1 * plane);
    _Float16* kbuf    = (_Float16*)(ws + 2 * plane);
    _Float16* vbuf    = (_Float16*)(ws + 3 * plane);
    _Float16* attnT   = (_Float16*)(ws + 4 * plane);
    _Float16* wqkv16  = (_Float16*)(ws + 5 * plane);                 // 1.5 MB
    _Float16* wproj16 = (_Float16*)((char*)wqkv16 + (size_t)3 * CH * CH * 2);

    cvt_kernel<<<(3 * CH * CH) / 256, 256, 0, stream>>>(w_qkv, wqkv16, 3 * CH * CH);
    cvt_kernel<<<(CH * CH) / 256, 256, 0, stream>>>(w_proj, wproj16, CH * CH);
    gn_kernel<<<BATCH * NGROUPS, 256, 0, stream>>>(x, gamma, beta, xT);
    qkv_kernel<<<dim3(12, 16, BATCH), 256, 0, stream>>>(xT, wqkv16, b_qkv,
                                                        qbuf, kbuf, vbuf);
    attn_kernel<<<512, 256, 0, stream>>>(qbuf, kbuf, vbuf, attnT);
    proj_kernel<<<dim3(4, 16, BATCH), 256, 0, stream>>>(attnT, wproj16, b_proj,
                                                        x, out);
}
